// TransformerAttention_17136919511749
// MI455X (gfx1250) — matrix operand order repaired and checked by reading, not compiled, not run on hardware
//
#include <hip/hip_runtime.h>
#include <hip/hip_bf16.h>

// ---------------------------------------------------------------------------
// Types / helpers
// ---------------------------------------------------------------------------
typedef __attribute__((ext_vector_type(16))) __bf16 v16bf;
typedef __attribute__((ext_vector_type(8)))  float  v8f;

union FragBF {
  v16bf bf;
  unsigned int u[8];
  uint4 q[2];
};

static __device__ __forceinline__ unsigned short f32_to_bf16(float f) {
  unsigned int u = __float_as_uint(f);
  u += 0x7FFFu + ((u >> 16) & 1u);   // round-to-nearest-even
  return (unsigned short)(u >> 16);
}

static __device__ __forceinline__ v8f v8f_zero() {
  v8f z;
#pragma unroll
  for (int i = 0; i < 8; ++i) z[i] = 0.0f;
  return z;
}

// LDS byte offset of a shared-memory pointer (flat LDS aperture: low 32 bits
// are the in-allocation offset per CDNA5 aperture mapping).
static __device__ __forceinline__ unsigned lds_off(const void* p) {
  return (unsigned)(unsigned long long)p;
}

// Async 16B copy global -> LDS (CDNA5 GLOBAL_LOAD_ASYNC_TO_LDS_B128, ASYNCcnt)
static __device__ __forceinline__ void async_copy_b128(unsigned lds,
                                                       const void* gptr) {
  unsigned long long g = (unsigned long long)gptr;
  asm volatile("global_load_async_to_lds_b128 %0, %1, off"
               :: "v"(lds), "v"(g) : "memory");
}

static __device__ __forceinline__ void wait_asynccnt0() {
  asm volatile("s_wait_asynccnt 0" ::: "memory");
}

#define S_LEN 2048
#define D_MODEL 1024
#define N_HEADS 16
#define DIM_HEAD 64

// ---------------------------------------------------------------------------
// fp32 -> bf16 conversion (weights)
// ---------------------------------------------------------------------------
__global__ void cvt_f32_bf16_kernel(const float* __restrict__ src,
                                    unsigned short* __restrict__ dst, int n) {
  int i = blockIdx.x * blockDim.x + threadIdx.x;
  int stride = gridDim.x * blockDim.x;
  for (; i < n; i += stride) dst[i] = f32_to_bf16(src[i]);
}

// ---------------------------------------------------------------------------
// LayerNorm (one row of 1024 per block) -> xn_bf16, plus raw x -> x_bf16
// ---------------------------------------------------------------------------
__global__ __launch_bounds__(256)
void ln_cvt_kernel(const float* __restrict__ x, const float* __restrict__ gamma,
                   unsigned short* __restrict__ xn_bf,
                   unsigned short* __restrict__ x_bf) {
  const int row = blockIdx.x;
  const int tid = threadIdx.x;
  const int lane = tid & 31;
  const int wv = tid >> 5;

  float4 xv = ((const float4*)(x + (size_t)row * D_MODEL))[tid];
  float s  = xv.x + xv.y + xv.z + xv.w;
  float s2 = xv.x * xv.x + xv.y * xv.y + xv.z * xv.z + xv.w * xv.w;
#pragma unroll
  for (int o = 1; o < 32; o <<= 1) {
    s  += __shfl_xor(s,  o, 32);
    s2 += __shfl_xor(s2, o, 32);
  }
  __shared__ float red[16];
  if (lane == 0) { red[wv] = s; red[8 + wv] = s2; }
  __syncthreads();
  float ts = 0.f, ts2 = 0.f;
#pragma unroll
  for (int i = 0; i < 8; ++i) { ts += red[i]; ts2 += red[8 + i]; }
  const float mu  = ts * (1.0f / D_MODEL);
  const float var = ts2 * (1.0f / D_MODEL) - mu * mu;
  const float inv = rsqrtf(var + 1e-5f);

  float4 gv = ((const float4*)gamma)[tid];
  const size_t base = (size_t)row * D_MODEL + tid * 4;
  xn_bf[base + 0] = f32_to_bf16((xv.x - mu) * inv * gv.x);
  xn_bf[base + 1] = f32_to_bf16((xv.y - mu) * inv * gv.y);
  xn_bf[base + 2] = f32_to_bf16((xv.z - mu) * inv * gv.z);
  xn_bf[base + 3] = f32_to_bf16((xv.w - mu) * inv * gv.w);
  x_bf[base + 0] = f32_to_bf16(xv.x);
  x_bf[base + 1] = f32_to_bf16(xv.y);
  x_bf[base + 2] = f32_to_bf16(xv.z);
  x_bf[base + 3] = f32_to_bf16(xv.w);
}

// ---------------------------------------------------------------------------
// Generic bf16 GEMM with fp32 accumulate: C[M,N] = scale * (A[M,K] @ B[K,N])
// Block tile 128x64, 8 waves, each wave computes 16x64 via 4 WMMA accumulators.
// A tile staged via GLOBAL_LOAD_ASYNC_TO_LDS_B128 (raw copy); B tile staged
// transposed through VGPRs (needs reshaping). Fragment gathers are contiguous
// dword LDS reads matching the CDNA5 WMMA VGPR layouts.
// ---------------------------------------------------------------------------
#define GT_M 128
#define GT_N 64
#define GT_K 32

template <int OUT_BF16>
__global__ __launch_bounds__(256)
void gemm_bf16_wmma_kernel(const unsigned short* __restrict__ A,
                           const unsigned short* __restrict__ Bm,
                           void* __restrict__ Cout,
                           int M, int N, int K, float scale) {
  __shared__ unsigned short sA[GT_M * GT_K];   // [128][32]
  __shared__ unsigned short sBt[GT_N * GT_K];  // [64][32] transposed

  const int tid  = threadIdx.x;
  const int lane = tid & 31;
  const int wv   = tid >> 5;        // 0..7 -> M sub-tile
  const int h    = (lane >> 4) & 1; // lane half
  const int ln   = lane & 15;
  const int m0 = blockIdx.y * GT_M;
  const int n0 = blockIdx.x * GT_N;

  v8f acc[4];
#pragma unroll
  for (int f = 0; f < 4; ++f) acc[f] = v8f_zero();

  for (int k0 = 0; k0 < K; k0 += GT_K) {
    // Stage A tile (128x32) with async global->LDS copies: 512 x 16B chunks,
    // 2 per thread.
    {
      int cid = tid;
#pragma unroll
      for (int rep = 0; rep < 2; ++rep, cid += 256) {
        const int r   = cid >> 2;
        const int seg = (cid & 3) << 3;  // element offset 0,8,16,24
        async_copy_b128(lds_off(&sA[r * GT_K + seg]),
                        A + (size_t)(m0 + r) * K + k0 + seg);
      }
    }
    // Stage B tile (32x64), transposed into sBt[n][k]
    {
      const int kk = tid >> 3;          // 0..31
      const int nn = (tid & 7) << 3;    // 0..56
      uint4 d = *(const uint4*)(Bm + (size_t)(k0 + kk) * N + n0 + nn);
      unsigned short tmp[8];
      *(uint4*)tmp = d;
#pragma unroll
      for (int j = 0; j < 8; ++j) sBt[(nn + j) * GT_K + kk] = tmp[j];
    }
    // Prefetch next K tile into caches while this one computes
    if (k0 + GT_K < K) {
      __builtin_prefetch(A + (size_t)(m0 + (tid >> 1)) * K + k0 + GT_K +
                             ((tid & 1) << 4), 0, 3);
      __builtin_prefetch(Bm + (size_t)(k0 + GT_K + (tid >> 3)) * N + n0 +
                              ((tid & 7) << 3), 0, 3);
    }
    wait_asynccnt0();
    __syncthreads();

    // A fragment: lane row = wv*16+ln, interleaved K layout
    FragBF a;
#pragma unroll
    for (int i = 0; i < 8; ++i) {
      const int kk = 2 * i + ((i >= 4) ? 8 : 0) + 8 * h;
      a.u[i] = *(const unsigned int*)(&sA[(wv * 16 + ln) * GT_K + kk]);
    }
#pragma unroll
    for (int f = 0; f < 4; ++f) {
      FragBF b;
#pragma unroll
      for (int i = 0; i < 8; ++i) {
        const int kk = 2 * i + ((i >= 4) ? 8 : 0) + 8 * h;
        b.u[i] = *(const unsigned int*)(&sBt[(f * 16 + ln) * GT_K + kk]);
      }
      acc[f] = __builtin_amdgcn_wmma_f32_16x16x32_bf16(
          false, a.bf, false, b.bf, (short)0, acc[f], false, false);
    }
    __syncthreads();
  }

  // Store C
#pragma unroll
  for (int f = 0; f < 4; ++f) {
#pragma unroll
    for (int v = 0; v < 8; ++v) {
      const size_t m = (size_t)(m0 + wv * 16 + v + 8 * h);
      const size_t n = (size_t)(n0 + f * 16 + ln);
      const float val = acc[f][v] * scale;
      if (OUT_BF16) ((unsigned short*)Cout)[m * N + n] = f32_to_bf16(val);
      else          ((float*)Cout)[m * N + n] = val;
    }
  }
}

// ---------------------------------------------------------------------------
// Flash attention (causal, single shared K/V head). One wave = one 16-row
// Q tile of one (batch, head). Steps 32 keys/iter: 4 sim WMMAs, online
// softmax, P round-trip through wave-private LDS, 4 PV WMMAs. V tile staged
// via async global->LDS copies.
// Q: [B*S, 1024] pre-scaled bf16.  KV: [B*S, 128] (k=cols 0..63, v=64..127).
// O: [B*S, 1024] bf16.
// ---------------------------------------------------------------------------
__global__ __launch_bounds__(128)
void flash_attn_wmma_kernel(const unsigned short* __restrict__ Q,
                            const unsigned short* __restrict__ KV,
                            unsigned short* __restrict__ O) {
  __shared__ unsigned short sP[4][16 * 32];       // per-wave P tile (16x32)
  __shared__ unsigned short sV[4][32 * DIM_HEAD]; // per-wave V tile (32x64)

  const int lane = threadIdx.x & 31;
  const int wv   = threadIdx.x >> 5;
  const int h    = (lane >> 4) & 1;
  const int ln   = lane & 15;
  const int it   = blockIdx.x * 4 + wv;  // q tile index 0..127
  const int head = blockIdx.y;
  const int b    = blockIdx.z;

  // Load Q fragments (A layout): two K-chunks of 32 covering DIM_HEAD=64
  FragBF qf[2];
  {
    const size_t qbase = ((size_t)(b * S_LEN + it * 16 + ln)) * D_MODEL + head * DIM_HEAD;
#pragma unroll
    for (int c = 0; c < 2; ++c) {
      const unsigned short* p = Q + qbase + c * 32 + 8 * h;
      qf[c].q[0] = *(const uint4*)(p);        // VGPR 0..3: K = 8h + 0..7
      qf[c].q[1] = *(const uint4*)(p + 16);   // VGPR 4..7: K = 8h + 16..23
    }
  }

  float mi[8], li[8];
  v8f acc[4];
#pragma unroll
  for (int v = 0; v < 8; ++v) { mi[v] = -1.0e30f; li[v] = 0.0f; }
#pragma unroll
  for (int f = 0; f < 4; ++f) acc[f] = v8f_zero();

  const int jt_max = (it * 16 + 15) >> 5;
  for (int jt = 0; jt <= jt_max; ++jt) {
    const int j0 = jt * 32;

    // Kick off async V tile staging (32 rows x 64 cols) early: lane moves one
    // 128B row as 8 x b128 async copies.
    {
      const unsigned vdst = lds_off(&sV[wv][lane * DIM_HEAD]);
      const unsigned short* vsrc =
          KV + (size_t)(b * S_LEN + j0 + lane) * 128 + 64;
#pragma unroll
      for (int u = 0; u < 8; ++u)
        async_copy_b128(vdst + u * 16, vsrc + u * 8);
    }

    // K fragments (B layout): 2 column sub-tiles x 2 K-chunks; contiguous reads
    FragBF kf[2][2];
#pragma unroll
    for (int s = 0; s < 2; ++s) {
      const unsigned short* krow = KV + (size_t)(b * S_LEN + j0 + s * 16 + ln) * 128;
#pragma unroll
      for (int c = 0; c < 2; ++c) {
        const unsigned short* p = krow + c * 32 + 8 * h;
        kf[s][c].q[0] = *(const uint4*)(p);
        kf[s][c].q[1] = *(const uint4*)(p + 16);
      }
    }

    // sim = (scaled Q) @ K^T for two 16x16 sub-tiles
    v8f sim[2];
#pragma unroll
    for (int s = 0; s < 2; ++s) {
      v8f z = v8f_zero();
      z = __builtin_amdgcn_wmma_f32_16x16x32_bf16(false, qf[0].bf, false, kf[s][0].bf,
                                                  (short)0, z, false, false);
      z = __builtin_amdgcn_wmma_f32_16x16x32_bf16(false, qf[1].bf, false, kf[s][1].bf,
                                                  (short)0, z, false, false);
      sim[s] = z;
    }

    // Causal mask
#pragma unroll
    for (int s = 0; s < 2; ++s) {
      const int j_g = j0 + s * 16 + ln;
#pragma unroll
      for (int v = 0; v < 8; ++v) {
        const int i_g = it * 16 + v + 8 * h;
        if (j_g > i_g) sim[s][v] = -3.0e38f;
      }
    }

    // Online softmax (per row: reduce across the 16-lane group)
    float alpha[8];
#pragma unroll
    for (int v = 0; v < 8; ++v) {
      float mx = fmaxf(sim[0][v], sim[1][v]);
#pragma unroll
      for (int o = 1; o < 16; o <<= 1) mx = fmaxf(mx, __shfl_xor(mx, o, 32));
      const float mnew = fmaxf(mi[v], mx);
      const float p0 = __expf(sim[0][v] - mnew);
      const float p1 = __expf(sim[1][v] - mnew);
      sim[0][v] = p0;
      sim[1][v] = p1;
      float rs = p0 + p1;
#pragma unroll
      for (int o = 1; o < 16; o <<= 1) rs += __shfl_xor(rs, o, 32);
      alpha[v] = __expf(mi[v] - mnew);
      li[v] = li[v] * alpha[v] + rs;
      mi[v] = mnew;
    }
#pragma unroll
    for (int f = 0; f < 4; ++f)
#pragma unroll
      for (int v = 0; v < 8; ++v) acc[f][v] *= alpha[v];

    // Write P (bf16) into wave-private LDS, row-major 16x32
#pragma unroll
    for (int s = 0; s < 2; ++s)
#pragma unroll
      for (int v = 0; v < 8; ++v)
        sP[wv][(v + 8 * h) * 32 + s * 16 + ln] = f32_to_bf16(sim[s][v]);

    // Wave-level LDS fence: ds stores (P) + async copies (V) feed the
    // cross-lane gathers below.
    __builtin_amdgcn_wave_barrier();
    wait_asynccnt0();
    asm volatile("s_wait_dscnt 0" ::: "memory");
    __builtin_amdgcn_wave_barrier();

    // Gather P as A fragment (16x32)
    FragBF pf;
#pragma unroll
    for (int i = 0; i < 8; ++i) {
      const int kk = 2 * i + ((i >= 4) ? 8 : 0) + 8 * h;
      pf.u[i] = *(const unsigned int*)(&sP[wv][ln * 32 + kk]);
    }
    // Gather V as 4 B fragments (32x16 each) and accumulate O += P @ V
#pragma unroll
    for (int f = 0; f < 4; ++f) {
      FragBF vf;
#pragma unroll
      for (int i = 0; i < 8; ++i) {
        const int kk = 2 * i + ((i >= 4) ? 8 : 0) + 8 * h;
        const unsigned int lo = sV[wv][kk * DIM_HEAD + f * 16 + ln];
        const unsigned int hi = sV[wv][(kk + 1) * DIM_HEAD + f * 16 + ln];
        vf.u[i] = lo | (hi << 16);
      }
      acc[f] = __builtin_amdgcn_wmma_f32_16x16x32_bf16(
          false, pf.bf, false, vf.bf, (short)0, acc[f], false, false);
    }
  }

  // Normalize and store (back to [B,S,H*Dh] layout, bf16)
#pragma unroll
  for (int v = 0; v < 8; ++v) li[v] = 1.0f / li[v];
#pragma unroll
  for (int f = 0; f < 4; ++f)
#pragma unroll
    for (int v = 0; v < 8; ++v) {
      const size_t row = (size_t)(b * S_LEN + it * 16 + v + 8 * h);
      O[row * D_MODEL + head * DIM_HEAD + f * 16 + ln] =
          f32_to_bf16(acc[f][v] * li[v]);
    }
}

// ---------------------------------------------------------------------------
// Host-side orchestration
// ---------------------------------------------------------------------------
extern "C" void kernel_launch(void* const* d_in, const int* in_sizes, int n_in,
                              void* d_out, int out_size, void* d_ws, size_t ws_size,
                              hipStream_t stream) {
  (void)in_sizes; (void)n_in; (void)out_size; (void)ws_size;
  const float* x     = (const float*)d_in[0];
  const float* gamma = (const float*)d_in[1];
  const float* Wq    = (const float*)d_in[2];
  const float* Wkv   = (const float*)d_in[3];
  const float* Wo    = (const float*)d_in[4];
  float* out = (float*)d_out;

  const int B = 2, S = S_LEN, D = D_MODEL;
  const int M = B * S;                 // 4096 rows
  const float SCALE = 0.125f;          // 64^-0.5

  // Workspace layout (bf16 buffers), 256B aligned
  char* ws = (char*)d_ws;
  size_t off = 0;
  auto take = [&](size_t bytes) -> unsigned short* {
    unsigned short* p = (unsigned short*)(ws + off);
    off += (bytes + 255) & ~(size_t)255;
    return p;
  };
  unsigned short* x_bf   = take((size_t)M * D * 2);
  unsigned short* xn_bf  = take((size_t)M * D * 2);
  unsigned short* Wq_bf  = take((size_t)D * 1024 * 2);
  unsigned short* Wkv_bf = take((size_t)D * 128 * 2);
  unsigned short* Wo_bf  = take((size_t)1024 * D * 2);
  unsigned short* q_bf   = take((size_t)M * 1024 * 2);
  unsigned short* kv_bf  = take((size_t)M * 128 * 2);
  unsigned short* ao_bf  = take((size_t)M * 1024 * 2);

  // 1) Weights -> bf16
  cvt_f32_bf16_kernel<<<512, 256, 0, stream>>>(Wq,  Wq_bf,  D * 1024);
  cvt_f32_bf16_kernel<<<512, 256, 0, stream>>>(Wkv, Wkv_bf, D * 128);
  cvt_f32_bf16_kernel<<<512, 256, 0, stream>>>(Wo,  Wo_bf,  1024 * D);

  // 2) LayerNorm + x -> bf16
  ln_cvt_kernel<<<M, 256, 0, stream>>>(x, gamma, xn_bf, x_bf);

  // 3) q = xn @ Wq, pre-scaled by SCALE (bf16 out)
  gemm_bf16_wmma_kernel<1><<<dim3(1024 / GT_N, M / GT_M), 256, 0, stream>>>(
      xn_bf, Wq_bf, (void*)q_bf, M, 1024, D, SCALE);

  // 4) kv = x @ Wkv (bf16 out); cols 0..63 = k, 64..127 = v
  gemm_bf16_wmma_kernel<1><<<dim3(128 / GT_N, M / GT_M), 256, 0, stream>>>(
      x_bf, Wkv_bf, (void*)kv_bf, M, 128, D, 1.0f);

  // 5) causal flash attention
  flash_attn_wmma_kernel<<<dim3(S / 16 / 4, N_HEADS, B), 128, 0, stream>>>(
      q_bf, kv_bf, ao_bf);

  // 6) out = ao @ Wo (fp32 out)
  gemm_bf16_wmma_kernel<0><<<dim3(1024 / GT_N, M / GT_M), 256, 0, stream>>>(
      ao_bf, Wo_bf, (void*)out, M, 1024, 1024, 1.0f);
}
